// SparseMultiHeadAttention_34497177321552
// MI455X (gfx1250) — matrix operand order repaired and checked by reading, not compiled, not run on hardware
//
#include <hip/hip_runtime.h>

#define SB 2
#define SEQ 2048
#define DM 1024
#define NH 16
#define DK 64
#define WIN 128
#define NGLOB 16

typedef _Float16 h8_t  __attribute__((ext_vector_type(8)));
typedef _Float16 h16_t __attribute__((ext_vector_type(16)));
typedef float    f8_t  __attribute__((ext_vector_type(8)));
typedef unsigned int u32x4 __attribute__((ext_vector_type(4)));
typedef unsigned int u32x8 __attribute__((ext_vector_type(8)));

__device__ __forceinline__ h16_t make_frag(h8_t a, h8_t b) {
  h16_t r;
#pragma unroll
  for (int i = 0; i < 8; ++i) { r[i] = a[i]; r[i + 8] = b[i]; }
  return r;
}

// A-fragment (16x32 f16, MxK): lane holds row (lane&15); 16 contiguous K values
// starting at (lane&16 ? 16 : 0), matching the B-fragment order.
__device__ __forceinline__ h16_t frag_ld_a(const _Float16* row) {
  const int lane = threadIdx.x & 31;
  const int lo = (lane & 16) ? 16 : 0;
  h8_t a = *(const h8_t*)(row + lo);
  h8_t b = *(const h8_t*)(row + lo + 8);
  return make_frag(a, b);
}

// B-fragment (32x16 f16, KxN): lane holds column (lane&15); 16 contiguous K values
// starting at (lane&16 ? 16 : 0).  `row` points at this column's 32 K-values.
__device__ __forceinline__ h16_t frag_ld_b(const _Float16* row) {
  const int lane = threadIdx.x & 31;
  const int ko = (lane & 16) ? 16 : 0;
  h8_t a = *(const h8_t*)(row + ko);
  h8_t b = *(const h8_t*)(row + ko + 8);
  return make_frag(a, b);
}

// Same fragments but sourced from fp32 LDS tiles (TDM cannot convert), with
// inline f32->f16 conversion.
__device__ __forceinline__ h16_t frag_ld_a_f32(const float* row) {
  const int lane = threadIdx.x & 31;
  const int lo = (lane & 16) ? 16 : 0;
  h16_t r;
#pragma unroll
  for (int i = 0; i < 8; ++i) {
    r[i] = (_Float16)row[lo + i];
    r[i + 8] = (_Float16)row[lo + 8 + i];
  }
  return r;
}
__device__ __forceinline__ h16_t frag_ld_b_f32(const float* row) {
  const int lane = threadIdx.x & 31;
  const int ko = (lane & 16) ? 16 : 0;
  h16_t r;
#pragma unroll
  for (int i = 0; i < 16; ++i) r[i] = (_Float16)row[ko + i];
  return r;
}

__device__ __forceinline__ f8_t wmma16x16x32(h16_t a, h16_t b, f8_t c) {
  return __builtin_amdgcn_wmma_f32_16x16x32_f16(false, a, false, b, (short)0, c,
                                                false, false);
}

// ---------------------------------------------------------------------------
// Tensor Data Mover: 2D fp32 tile (tile_d0 x tile_d1 elements) global -> LDS,
// with LDS row padding (32 data DWORDs + 4 pad DWORDs => 36-DWORD rows).
// Descriptor packing per CDNA5 ISA 8.3/8.4 (count=1, type=2, data_size=4B).
// ---------------------------------------------------------------------------
__device__ __forceinline__ void tdm_load_2d_f32(void* lds_ptr, const void* gptr,
                                                unsigned tensor_d0,
                                                unsigned tensor_d1,
                                                unsigned tile_d0,
                                                unsigned tile_d1,
                                                unsigned stride_d0) {
  const unsigned lds_off = (unsigned)(unsigned long long)lds_ptr;
  const unsigned long long ga = (unsigned long long)gptr;
  u32x4 g0;
  g0[0] = 1u;                                   // count=1, user descriptor
  g0[1] = lds_off;                              // lds_addr
  g0[2] = (unsigned)ga;                         // global_addr[31:0]
  g0[3] = (unsigned)((ga >> 32) & 0x01FFFFFFu)  // global_addr[56:32]
          | (2u << 30);                         // type=2 (image)
  u32x8 g1;
  g1[0] = (2u << 16)                            // data_size = 4 bytes
          | (1u << 20)                          // pad_enable
          | (4u << 22)                          // pad_interval: 32 DWORDs
          | (3u << 25);                         // pad_amount:   4 DWORDs
  g1[1] = (tensor_d0 & 0xFFFFu) << 16;          // tensor_dim0[15:0]
  g1[2] = (tensor_d0 >> 16) | ((tensor_d1 & 0xFFFFu) << 16);
  g1[3] = (tensor_d1 >> 16) | (tile_d0 << 16);  // tile_dim0
  g1[4] = tile_d1;                              // tile_dim1 (tile_dim2=0)
  g1[5] = stride_d0;                            // tensor_dim0_stride[31:0]
  g1[6] = 0u;
  g1[7] = 0u;
  u32x4 gz;
  gz[0] = 0u; gz[1] = 0u; gz[2] = 0u; gz[3] = 0u;
  asm volatile("tensor_load_to_lds %0, %1, %2, %3"
               :
               : "s"(g0), "s"(g1), "s"(gz), "s"(gz)
               : "memory");
}

// ---------------------------------------------------------------------------
// Kernel 1: Y = X @ W^T + b for Q/K/V (blockIdx.z selects), + RoPE, -> f16.
// 64(M) x 64(N) workgroup tile; K stepped by 32 with TDM-staged, double-
// buffered fp32 LDS tiles.  4 waves: 16-wide N slab x 4 M subtiles each.
// ---------------------------------------------------------------------------
__global__ __launch_bounds__(128) void qkv_rope_kernel(
    const float* __restrict__ qin, const float* __restrict__ kin,
    const float* __restrict__ vin, const float* __restrict__ wq,
    const float* __restrict__ wk, const float* __restrict__ wv,
    const float* __restrict__ bq, const float* __restrict__ bk,
    const float* __restrict__ bv, _Float16* __restrict__ Qh,
    _Float16* __restrict__ Kh, _Float16* __restrict__ Vt) {
  __shared__ float As[2][64][36];   // 32 data + 4 pad DWORDs per row (TDM pad)
  __shared__ float Bs[2][64][36];
  __shared__ float Cs[64][64];

  const int z = blockIdx.z;
  const float* X = (z == 0) ? qin : (z == 1) ? kin : vin;
  const float* W = (z == 0) ? wq : (z == 1) ? wk : wv;
  const float* bias = (z == 0) ? bq : (z == 1) ? bk : bv;

  const int n0 = blockIdx.x * 64;
  const int m0 = blockIdx.y * 64;
  const int tid = threadIdx.x;
  const int wave = tid >> 5;
  const int lane = tid & 31;
  const int lr = lane & 15;
  const int hi = lane >> 4;

  const float* Xt = X + (size_t)m0 * DM;  // tile row base; k offset added below
  const float* Wt = W + (size_t)n0 * DM;

  f8_t acc[4];
#pragma unroll
  for (int t = 0; t < 4; ++t)
#pragma unroll
    for (int r = 0; r < 8; ++r) acc[t][r] = 0.f;

  // prologue: DMA first K-slab into buffer 0
  if (wave == 0)
    tdm_load_2d_f32(&As[0][0][0], Xt, DM, SB * SEQ, 32, 64, DM);
  else if (wave == 1)
    tdm_load_2d_f32(&Bs[0][0][0], Wt, DM, DM, 32, 64, DM);
  if (wave < 2) __builtin_amdgcn_s_wait_tensorcnt(0);
  __syncthreads();

  for (int k0 = 0; k0 < DM; k0 += 32) {
    const int buf = (k0 >> 5) & 1;
    if (k0 + 32 < DM) {  // DMA next K-slab into the other buffer
      if (wave == 0)
        tdm_load_2d_f32(&As[buf ^ 1][0][0], Xt + k0 + 32, DM, SB * SEQ, 32, 64,
                        DM);
      else if (wave == 1)
        tdm_load_2d_f32(&Bs[buf ^ 1][0][0], Wt + k0 + 32, DM, DM, 32, 64, DM);
    }
    h16_t bf = frag_ld_b_f32(&Bs[buf][wave * 16 + lr][0]);
#pragma unroll
    for (int mt = 0; mt < 4; ++mt) {
      h16_t af = frag_ld_a_f32(&As[buf][mt * 16 + lr][0]);
      acc[mt] = wmma16x16x32(af, bf, acc[mt]);
    }
    if (wave < 2) __builtin_amdgcn_s_wait_tensorcnt(0);
    __syncthreads();
  }

  const float bval = bias[n0 + wave * 16 + lr];
#pragma unroll
  for (int mt = 0; mt < 4; ++mt)
#pragma unroll
    for (int r = 0; r < 8; ++r)
      Cs[mt * 16 + r + 8 * hi][wave * 16 + lr] = acc[mt][r] + bval;
  __syncthreads();

  // epilogue: bias already added; apply RoPE (z<2) and store f16 head-major.
  const int row = tid >> 1;
  const int gm = m0 + row;
  const int bb = gm >> 11;        // batch
  const int ss = gm & (SEQ - 1);  // seq position
  const int head = blockIdx.x;    // n0 / 64 (tile width == head width)
  const int c0 = (tid & 1) * 32;

  if (z < 2) {
    _Float16* dst =
        ((z == 0) ? Qh : Kh) + (((size_t)(bb * NH + head)) * SEQ + ss) * DK;
    const float qscale = (z == 0) ? 0.125f : 1.0f;  // fold 1/sqrt(dk) into Q
#pragma unroll
    for (int d = c0; d < c0 + 32; ++d) {
      float xv = Cs[row][d];
      float xr = (d < 32) ? -Cs[row][2 * d + 1] : Cs[row][2 * (d - 32)];
      float invf = __powf(10000.f, -(float)(d & 31) * (1.f / 32.f));
      float ang = (float)ss * invf;
      float sn, cn;
      __sincosf(ang, &sn, &cn);
      dst[d] = (_Float16)((xv * cn + xr * sn) * qscale);
    }
  } else {
    // V stored transposed: Vt[(bh*64 + d)*SEQ + s]
    _Float16* dst = Vt + ((size_t)(bb * NH + head)) * DK * SEQ + ss;
#pragma unroll
    for (int d = c0; d < c0 + 32; ++d)
      dst[(size_t)d * SEQ] = (_Float16)Cs[row][d];
  }
}

// ---------------------------------------------------------------------------
// Kernel 2: windowed/global causal flash attention; one wave per 16-query tile.
// ---------------------------------------------------------------------------
__global__ __launch_bounds__(128) void attn_kernel(
    const _Float16* __restrict__ Qh, const _Float16* __restrict__ Kh,
    const _Float16* __restrict__ Vt, _Float16* __restrict__ AO) {
  __shared__ _Float16 Pls[4][16][40];

  const int tid = threadIdx.x;
  const int wave = tid >> 5;
  const int lane = tid & 31;
  const int lr = lane & 15;
  const int hi = lane >> 4;

  const int wid = blockIdx.x * 4 + wave;  // 0..4095
  const int qt = wid & (SEQ / 16 - 1);    // query tile 0..127
  const int bh = wid >> 7;                // 0..31
  const int bb = bh >> 4;
  const int hh = bh & 15;
  const int i0 = qt * 16;

  const _Float16* Qp = Qh + (size_t)bh * SEQ * DK;
  const _Float16* Kp = Kh + (size_t)bh * SEQ * DK;
  const _Float16* Vp = Vt + (size_t)bh * DK * SEQ;

  h16_t qf0 = frag_ld_a(Qp + (size_t)(i0 + lr) * DK + 0);
  h16_t qf1 = frag_ld_a(Qp + (size_t)(i0 + lr) * DK + 32);

  float m_run[8], l_run[8];
  f8_t oacc[4];
#pragma unroll
  for (int r = 0; r < 8; ++r) { m_run[r] = -1e30f; l_run[r] = 0.f; }
#pragma unroll
  for (int t = 0; t < 4; ++t)
#pragma unroll
    for (int r = 0; r < 8; ++r) oacc[t][r] = 0.f;

  // rows < 16 are global and attend to everything -> tile 0 sweeps all keys.
  const int jmax = (i0 == 0) ? (SEQ - 1) : (i0 + 15);
  int wlo = i0 - WIN;
  wlo = (wlo < 32) ? 32 : (wlo & ~31);

  for (int j0 = 0; j0 <= jmax; j0 = (j0 == 0) ? wlo : (j0 + 32)) {
    // prefetch next chunk's K/V rows into cache
    const int jn = (j0 == 0) ? wlo : (j0 + 32);
    if (jn <= jmax) {
      __builtin_prefetch((const void*)(Kp + (size_t)(jn + lr) * DK), 0, 1);
      __builtin_prefetch((const void*)(Vp + (size_t)lr * SEQ + jn), 0, 1);
    }
    // scores: 16 x 32 chunk = two 16x16 C tiles, 2 WMMAs each (dk=64)
    f8_t sc[2];
#pragma unroll
    for (int jt = 0; jt < 2; ++jt) {
      f8_t s;
#pragma unroll
      for (int r = 0; r < 8; ++r) s[r] = 0.f;
      const _Float16* krow = Kp + (size_t)(j0 + jt * 16 + lr) * DK;
      s = wmma16x16x32(qf0, frag_ld_b(krow), s);
      s = wmma16x16x32(qf1, frag_ld_b(krow + 32), s);
      sc[jt] = s;
    }
    // mask + per-row max (rows live in vgpr index, cols across 16-lane groups)
    float cmax[8];
#pragma unroll
    for (int r = 0; r < 8; ++r) cmax[r] = -1e30f;
#pragma unroll
    for (int jt = 0; jt < 2; ++jt)
#pragma unroll
      for (int r = 0; r < 8; ++r) {
        int i = i0 + r + 8 * hi;
        int j = j0 + jt * 16 + lr;
        bool ok = (i < NGLOB) || (j < NGLOB) || ((j <= i) && (i - j <= WIN));
        float v = ok ? sc[jt][r] : -1e30f;
        sc[jt][r] = v;
        cmax[r] = fmaxf(cmax[r], v);
      }
#pragma unroll
    for (int off = 1; off < 16; off <<= 1)
#pragma unroll
      for (int r = 0; r < 8; ++r)
        cmax[r] = fmaxf(cmax[r], __shfl_xor(cmax[r], off, 32));

    float rscale[8];
#pragma unroll
    for (int r = 0; r < 8; ++r) {
      float mnew = fmaxf(m_run[r], cmax[r]);
      rscale[r] = __expf(m_run[r] - mnew);
      m_run[r] = mnew;
      l_run[r] *= rscale[r];
    }
    // exp, stage P into per-wave LDS slab (C-layout -> A-layout redistribution)
    float psum[8];
#pragma unroll
    for (int r = 0; r < 8; ++r) psum[r] = 0.f;
#pragma unroll
    for (int jt = 0; jt < 2; ++jt)
#pragma unroll
      for (int r = 0; r < 8; ++r) {
        float p = __expf(sc[jt][r] - m_run[r]);
        psum[r] += p;
        Pls[wave][r + 8 * hi][jt * 16 + lr] = (_Float16)p;
      }
#pragma unroll
    for (int off = 1; off < 16; off <<= 1)
#pragma unroll
      for (int r = 0; r < 8; ++r) psum[r] += __shfl_xor(psum[r], off, 32);
#pragma unroll
    for (int r = 0; r < 8; ++r) l_run[r] += psum[r];

#pragma unroll
    for (int t = 0; t < 4; ++t)
#pragma unroll
      for (int r = 0; r < 8; ++r) oacc[t][r] *= rscale[r];

    // P(16x32) @ V(32x64): 4 WMMAs, V columns contiguous via transposed layout
    h16_t pf = frag_ld_a(&Pls[wave][lr][0]);
#pragma unroll
    for (int t = 0; t < 4; ++t) {
      const _Float16* vrow = Vp + (size_t)(t * 16 + lr) * SEQ + j0;
      oacc[t] = wmma16x16x32(pf, frag_ld_b(vrow), oacc[t]);
    }
  }

#pragma unroll
  for (int r = 0; r < 8; ++r) l_run[r] = 1.f / l_run[r];
#pragma unroll
  for (int t = 0; t < 4; ++t)
#pragma unroll
    for (int r = 0; r < 8; ++r) {
      int i = i0 + r + 8 * hi;
      AO[((size_t)bb * SEQ + i) * DM + hh * DK + t * 16 + lr] =
          (_Float16)(oacc[t][r] * l_run[r]);
    }
}

// ---------------------------------------------------------------------------
// Kernel 3: out = AO @ Wo^T + bo  (fp32 out). Direct-from-global fragments.
// One wave per 64(M) x 16(N) tile.
// ---------------------------------------------------------------------------
__global__ __launch_bounds__(256) void out_proj_kernel(
    const _Float16* __restrict__ AO, const float* __restrict__ Wo,
    const float* __restrict__ bo, float* __restrict__ out) {
  const int tid = threadIdx.x;
  const int lane = tid & 31;
  const int lr = lane & 15;
  const int hi = lane >> 4;
  const int wid = blockIdx.x * 8 + (tid >> 5);  // 0..4095
  const int nt = wid & 63;
  const int mt = wid >> 6;
  const int n0 = nt * 16;
  const int m0 = mt * 64;

  f8_t acc[4];
#pragma unroll
  for (int t = 0; t < 4; ++t)
#pragma unroll
    for (int r = 0; r < 8; ++r) acc[t][r] = 0.f;

  const int ko = (lane & 16) ? 16 : 0;
  for (int k0 = 0; k0 < DM; k0 += 32) {
    const float* wr = Wo + (size_t)(n0 + lr) * DM + k0 + ko;
    if (k0 + 32 < DM) {
      __builtin_prefetch((const void*)(wr + 32), 0, 1);
      __builtin_prefetch(
          (const void*)(AO + (size_t)(m0 + lr) * DM + k0 + 32), 0, 1);
    }
    h16_t bf;
#pragma unroll
    for (int i = 0; i < 8; ++i) {
      bf[i] = (_Float16)wr[i];
      bf[i + 8] = (_Float16)wr[i + 8];
    }
#pragma unroll
    for (int t = 0; t < 4; ++t) {
      const _Float16* ar = AO + (size_t)(m0 + t * 16 + lr) * DM + k0;
      acc[t] = wmma16x16x32(frag_ld_a(ar), bf, acc[t]);
    }
  }
  const float bval = bo[n0 + lr];
#pragma unroll
  for (int t = 0; t < 4; ++t)
#pragma unroll
    for (int r = 0; r < 8; ++r) {
      int m = m0 + t * 16 + r + 8 * hi;
      out[(size_t)m * DM + n0 + lr] = acc[t][r] + bval;
    }
}

extern "C" void kernel_launch(void* const* d_in, const int* in_sizes, int n_in,
                              void* d_out, int out_size, void* d_ws,
                              size_t ws_size, hipStream_t stream) {
  (void)in_sizes; (void)n_in; (void)out_size; (void)ws_size;
  const float* q  = (const float*)d_in[0];
  const float* k  = (const float*)d_in[1];
  const float* v  = (const float*)d_in[2];
  const float* wq = (const float*)d_in[3];
  const float* bq = (const float*)d_in[4];
  const float* wk = (const float*)d_in[5];
  const float* bk = (const float*)d_in[6];
  const float* wv = (const float*)d_in[7];
  const float* bv = (const float*)d_in[8];
  const float* wo = (const float*)d_in[9];
  const float* bo = (const float*)d_in[10];
  float* out = (float*)d_out;

  char* ws = (char*)d_ws;
  _Float16* Qh = (_Float16*)(ws);                              // 8 MB
  _Float16* Kh = (_Float16*)(ws + (size_t)8 * 1024 * 1024);    // 8 MB
  _Float16* Vt = (_Float16*)(ws + (size_t)16 * 1024 * 1024);   // 8 MB
  _Float16* AO = (_Float16*)(ws + (size_t)24 * 1024 * 1024);   // 8 MB

  qkv_rope_kernel<<<dim3(16, 64, 3), 128, 0, stream>>>(
      q, k, v, wq, wk, wv, bq, bk, bv, Qh, Kh, Vt);
  attn_kernel<<<dim3(1024), 128, 0, stream>>>(Qh, Kh, Vt, AO);
  out_proj_kernel<<<dim3(512), 256, 0, stream>>>(AO, wo, bo, out);
}
